// LeNet5_CSNN_63513976373967
// MI455X (gfx1250) — matrix operand order repaired and checked by reading, not compiled, not run on hardware
//
#include <hip/hip_runtime.h>

// ---------------------------------------------------------------------------
// Spiking LeNet (CSNN) forward on gfx1250.
//   conv1 (K=50): direct f32 VALU (tiny).
//   conv2 (K=800), fc1 (K=3136), fc2 (K=128): f16 implicit-GEMM via
//   v_wmma_f32_16x16x32_f16, activations NHWC so K is contiguous.
//   conv2 input carries a 2-pixel zero halo so all im2col loads are
//   branch-free; B fragments use distinct registers + next-A prefetch so
//   load waits stagger (loadcnt 6/4/2/0) instead of serializing on 0.
//   Pre-BN conv outputs stored f16 (stats accumulated in f32); all
//   intermediates (~133 MB) fit in the 192 MB L2.
// ---------------------------------------------------------------------------

#define BTT   1280          // B*T = 64*20
#define TSTEP 20
#define LBETA 0.9f
#define BNEPS 1e-5f

typedef __attribute__((ext_vector_type(16))) _Float16 v16h;
typedef __attribute__((ext_vector_type(8)))  _Float16 v8h;
typedef __attribute__((ext_vector_type(8)))  float    v8f;

union FragH { v16h v; v8h h[2]; };

__device__ __forceinline__ v8f zero_v8f() {
  v8f z;
#pragma unroll
  for (int i = 0; i < 8; ++i) z[i] = 0.f;
  return z;
}

// ---------------- workspace layout (bytes, 256-aligned) --------------------
#define OFF_Z1    0ull            // [1280][28][28][32] f16   64,225,280 B
#define OFF_Z2    64225280ull     // [1280][14][14][64] f16   32,112,640 B
#define OFF_Z3    96337920ull     // [1280][128]        f32      655,360 B
#define OFF_Z4    96993280ull     // [1280][16]         f32       81,920 B
#define OFF_A1    97075200ull     // [1280][18][18][32] f16 halo-2  26,542,080 B
#define OFF_A2    123617280ull    // [1280][7][7][64]   f16    8,028,160 B
#define OFF_S3    131645440ull    // [1280][128]        f16      327,680 B
#define OFF_W2H   131973120ull    // [64][800]          f16      102,400 B
#define OFF_WFC1H 132075520ull    // [128][3136]        f16      802,816 B
#define OFF_WFC2H 132878336ull    // [16][128]          f16        4,096 B
#define OFF_STATS 132882432ull    // stats1 (64 f32) + stats2 (128 f32)
#define A1_BYTES  26542080ull

// ---------------------------------------------------------------------------
// Weight repack to f16 with GEMM-friendly K order.
// ---------------------------------------------------------------------------
__global__ void repack_weights(const float* __restrict__ w2,
                               const float* __restrict__ wfc1,
                               const float* __restrict__ wfc2,
                               _Float16* __restrict__ w2h,
                               _Float16* __restrict__ wfc1h,
                               _Float16* __restrict__ wfc2h) {
  const int N1 = 64 * 800, N2 = 128 * 3136, N3 = 16 * 128;
  int id = blockIdx.x * blockDim.x + threadIdx.x;
  if (id < N1) {
    int n = id / 800, k = id % 800;
    int pos = k >> 5, ci = k & 31;
    int kh = pos / 5, kw = pos % 5;
    w2h[id] = (_Float16)w2[((n * 32 + ci) * 5 + kh) * 5 + kw];
  } else if (id < N1 + N2) {
    int j = id - N1;
    int n = j / 3136, k = j % 3136;
    int hw = k >> 6, c = k & 63;
    wfc1h[j] = (_Float16)wfc1[n * 3136 + c * 49 + hw];
  } else if (id < N1 + N2 + N3) {
    int j = id - N1 - N2;
    int n = j >> 7, k = j & 127;
    wfc2h[j] = (n < 10) ? (_Float16)wfc2[n * 128 + k] : (_Float16)0.f;
  }
}

// ---------------------------------------------------------------------------
// conv1: 2->32 ch, 5x5 pad2, direct f32, f16 NHWC out.
// ---------------------------------------------------------------------------
__global__ void conv1_direct(const float* __restrict__ x,
                             const float* __restrict__ w1,
                             _Float16* __restrict__ z1) {
  int id = blockIdx.x * blockDim.x + threadIdx.x;   // 1280*784*32 exact
  int c = id & 31;
  int q = id >> 5;
  int ow = q % 28; q /= 28;
  int oh = q % 28;
  int bt = q / 28;
  const float* xb = x + (size_t)bt * 2 * 784;
  const float* wb = w1 + c * 50;
  float acc = 0.f;
#pragma unroll
  for (int ci = 0; ci < 2; ++ci) {
#pragma unroll
    for (int kh = 0; kh < 5; ++kh) {
      int ih = oh + kh - 2;
      if (ih < 0 || ih >= 28) continue;
#pragma unroll
      for (int kw = 0; kw < 5; ++kw) {
        int iw = ow + kw - 2;
        if (iw < 0 || iw >= 28) continue;
        acc += xb[ci * 784 + ih * 28 + iw] * wb[ci * 25 + kh * 5 + kw];
      }
    }
  }
  z1[((size_t)bt * 784 + oh * 28 + ow) * 32 + c] = (_Float16)acc;
}

// ---------------------------------------------------------------------------
// BN batch stats from f16 NHWC tensor, f32 accumulation.
// ---------------------------------------------------------------------------
__global__ void bn_stats(const _Float16* __restrict__ z, int npix, int C,
                         float* __restrict__ stats) {
  __shared__ float ssum[64];
  __shared__ float ssq[64];
  int tid = threadIdx.x;
  if (tid < C) { ssum[tid] = 0.f; ssq[tid] = 0.f; }
  __syncthreads();
  int c = tid % C;
  int rows = blockDim.x / C;
  int row = tid / C;
  float s = 0.f, q = 0.f;
  for (int p = blockIdx.x * rows + row; p < npix; p += gridDim.x * rows) {
    float v = (float)z[(size_t)p * C + c];
    s += v; q += v * v;
  }
  atomicAdd(&ssum[c], s);
  atomicAdd(&ssq[c], q);
  __syncthreads();
  if (tid < C) {
    atomicAdd(&stats[tid], ssum[tid]);
    atomicAdd(&stats[C + tid], ssq[tid]);
  }
}

// ---------------------------------------------------------------------------
// Fused BN-apply + LIF (T=20) + 2x2 avg-pool.  Thread = (b, ph, pw, c).
// Output NHWC f16, optionally with a zero border (halo for next conv).
// ---------------------------------------------------------------------------
__global__ void lif_pool(const _Float16* __restrict__ z,   // [BT][H][H][C]
                         const float* __restrict__ stats,
                         const float* __restrict__ gamma,
                         const float* __restrict__ beta,
                         _Float16* __restrict__ a,         // [BT][oW][oW][C]
                         int H, int C, int Cmask, int Cshift, float invcnt,
                         int outW, int border) {
  int id = blockIdx.x * blockDim.x + threadIdx.x;
  int c = id & Cmask;
  int q = id >> Cshift;
  int P = H >> 1;
  int pw = q % P; q /= P;
  int ph = q % P;
  int b  = q / P;
  float mean  = stats[c] * invcnt;
  float var   = stats[C + c] * invcnt - mean * mean;
  float scale = gamma[c] * rsqrtf(var + BNEPS);
  float shift = beta[c] - mean * scale;
  int ih = 2 * ph, iw = 2 * pw;
  size_t row0 = (size_t)(ih * H + iw) * C + c;
  size_t row1 = row0 + (size_t)H * C;
  size_t img  = (size_t)H * H * C;
  size_t pimg = (size_t)outW * outW * C;
  size_t oidx = (size_t)((ph + border) * outW + (pw + border)) * C + c;
  float v0 = 0.f, v1 = 0.f, v2 = 0.f, v3 = 0.f;
  for (int t = 0; t < TSTEP; ++t) {
    size_t base = (size_t)(b * TSTEP + t) * img;
    float x0 = (float)z[base + row0]     * scale + shift;
    float x1 = (float)z[base + row0 + C] * scale + shift;
    float x2 = (float)z[base + row1]     * scale + shift;
    float x3 = (float)z[base + row1 + C] * scale + shift;
    v0 = v0 * LBETA + x0; float s0 = (v0 >= 1.f) ? 1.f : 0.f; v0 -= s0;
    v1 = v1 * LBETA + x1; float s1 = (v1 >= 1.f) ? 1.f : 0.f; v1 -= s1;
    v2 = v2 * LBETA + x2; float s2 = (v2 >= 1.f) ? 1.f : 0.f; v2 -= s2;
    v3 = v3 * LBETA + x3; float s3 = (v3 >= 1.f) ? 1.f : 0.f; v3 -= s3;
    a[(size_t)(b * TSTEP + t) * pimg + oidx] =
        (_Float16)((s0 + s1 + s2 + s3) * 0.25f);
  }
}

// ---------------------------------------------------------------------------
// conv2 as implicit GEMM via WMMA, software-pipelined.
// Input has 2-pixel zero halo: all loads unconditional.
// M = 196 pixels (13 tiles of 16), N = 64 (4 tiles), K = 800 (25 x 32ci).
// Per (kh,kw): 4 distinct B fragments + prefetched next-A fragment so the
// scheduler can stagger s_wait_loadcnt across the 4 WMMAs.
// ---------------------------------------------------------------------------
__global__ void conv2_wmma(const _Float16* __restrict__ a1h, // [1280][18][18][32]
                           const _Float16* __restrict__ w2h, // [64][800]
                           _Float16* __restrict__ z2) {      // [1280][14][14][64]
  int wave = threadIdx.x >> 5;
  int lane = threadIdx.x & 31;
  int tile = blockIdx.x * 4 + wave;          // 0..16639
  int img  = tile / 13;
  int mt   = tile % 13;
  int mlo  = lane & 15;
  int half = lane >> 4;
  int ko   = half * 16;
  int p    = mt * 16 + mlo;
  int pc   = (p < 196) ? p : 195;            // clamp: halo makes loads safe
  int oh = pc / 14, ow = pc % 14;

  const _Float16* abase = a1h + ((size_t)img * 324 + (size_t)oh * 18 + ow) * 32;
  const _Float16* bbase = w2h + (size_t)mlo * 800 + half * 16;

  v8f acc[4];
#pragma unroll
  for (int i = 0; i < 4; ++i) acc[i] = zero_v8f();

  // prologue: A fragment for (kh=0, kw=0)
  FragH af;
  af.h[0] = *(const v8h*)(abase + ko);
  af.h[1] = *(const v8h*)(abase + 8 + ko);

  for (int kh = 0; kh < 5; ++kh) {
    const _Float16* arow = abase + (size_t)kh * (18 * 32);
    const _Float16* brow = bbase + (size_t)kh * (5 * 32);
#pragma unroll
    for (int kw = 0; kw < 5; ++kw) {
      const _Float16* bp = brow + kw * 32;
      FragH b0, b1, b2, b3;                 // distinct regs -> staggered waits
      b0.v = *(const v16h*)(bp);
      b1.v = *(const v16h*)(bp + 12800);
      b2.v = *(const v16h*)(bp + 25600);
      b3.v = *(const v16h*)(bp + 38400);
      // prefetch next A fragment (next kw, or next kh row; clamped at end)
      const _Float16* an;
      if (kw < 4) an = arow + (kw + 1) * 32;
      else        an = abase + (size_t)((kh < 4) ? kh + 1 : kh) * (18 * 32);
      FragH afn;
      afn.h[0] = *(const v8h*)(an + ko);
      afn.h[1] = *(const v8h*)(an + 8 + ko);

      acc[0] = __builtin_amdgcn_wmma_f32_16x16x32_f16(
          false, af.v, false, b0.v, (short)0, acc[0], false, false);
      acc[1] = __builtin_amdgcn_wmma_f32_16x16x32_f16(
          false, af.v, false, b1.v, (short)0, acc[1], false, false);
      acc[2] = __builtin_amdgcn_wmma_f32_16x16x32_f16(
          false, af.v, false, b2.v, (short)0, acc[2], false, false);
      acc[3] = __builtin_amdgcn_wmma_f32_16x16x32_f16(
          false, af.v, false, b3.v, (short)0, acc[3], false, false);
      af = afn;
    }
  }

  int mrow = mt * 16 + half * 8;
  size_t ob = (size_t)img * 196;
  if (mt != 12) {                              // full tile: branchless stores
#pragma unroll
    for (int nt = 0; nt < 4; ++nt)
#pragma unroll
      for (int r = 0; r < 8; ++r)
        z2[(ob + mrow + r) * 64 + nt * 16 + mlo] = (_Float16)acc[nt][r];
  } else {
#pragma unroll
    for (int nt = 0; nt < 4; ++nt)
#pragma unroll
      for (int r = 0; r < 8; ++r)
        if (mrow + r < 196)
          z2[(ob + mrow + r) * 64 + nt * 16 + mlo] = (_Float16)acc[nt][r];
  }
}

// ---------------------------------------------------------------------------
// fc1: [1280][3136] @ [3136][128] via WMMA, 2-deep double buffer.
// 80 M-tiles x 8 N-tiles = 640 waves; K = 98 chunks of 32 (even count).
// ---------------------------------------------------------------------------
__global__ void fc1_wmma(const _Float16* __restrict__ a2,    // [1280][3136]
                         const _Float16* __restrict__ wfc1h, // [128][3136]
                         float* __restrict__ z3) {           // [1280][128]
  int wave = threadIdx.x >> 5;
  int lane = threadIdx.x & 31;
  int g  = blockIdx.x * 4 + wave;   // 0..639
  int mt = g >> 3;
  int nt = g & 7;
  int mlo = lane & 15, half = lane >> 4, ko = half * 16;
  const _Float16* arow = a2 + (size_t)(mt * 16 + mlo) * 3136;
  const _Float16* brow = wfc1h + (size_t)(nt * 16 + mlo) * 3136;
  v8f acc = zero_v8f();

  FragH a0, b0;
  a0.h[0] = *(const v8h*)(arow + ko);
  a0.h[1] = *(const v8h*)(arow + 8 + ko);
  b0.v    = *(const v16h*)(brow + half * 16);

  for (int kc = 0; kc < 98; kc += 2) {
    int kb1 = (kc + 1) * 32;
    FragH a1, b1;
    a1.h[0] = *(const v8h*)(arow + kb1 + ko);
    a1.h[1] = *(const v8h*)(arow + kb1 + 8 + ko);
    b1.v    = *(const v16h*)(brow + kb1 + half * 16);
    acc = __builtin_amdgcn_wmma_f32_16x16x32_f16(
        false, a0.v, false, b0.v, (short)0, acc, false, false);

    int kb2 = ((kc + 2 < 98) ? kc + 2 : kc) * 32;   // clamped tail reload
    a0.h[0] = *(const v8h*)(arow + kb2 + ko);
    a0.h[1] = *(const v8h*)(arow + kb2 + 8 + ko);
    b0.v    = *(const v16h*)(brow + kb2 + half * 16);
    acc = __builtin_amdgcn_wmma_f32_16x16x32_f16(
        false, a1.v, false, b1.v, (short)0, acc, false, false);
  }
#pragma unroll
  for (int r = 0; r < 8; ++r)
    z3[(size_t)(mt * 16 + half * 8 + r) * 128 + nt * 16 + mlo] = acc[r];
}

// ---------------------------------------------------------------------------
// LIF over fc1 output (no BN): thread = (b, j<128).
// ---------------------------------------------------------------------------
__global__ void lif_fc1(const float* __restrict__ z3, _Float16* __restrict__ s3) {
  int id = blockIdx.x * blockDim.x + threadIdx.x;  // 8192 exact
  int j = id & 127;
  int b = id >> 7;
  float v = 0.f;
  for (int t = 0; t < TSTEP; ++t) {
    float x = z3[(size_t)(b * TSTEP + t) * 128 + j];
    v = v * LBETA + x;
    float s = (v >= 1.f) ? 1.f : 0.f;
    v -= s;
    s3[(size_t)(b * TSTEP + t) * 128 + j] = (_Float16)s;
  }
}

// ---------------------------------------------------------------------------
// fc2: [1280][128] @ [128][16] (N padded 10->16).  80 waves.
// ---------------------------------------------------------------------------
__global__ void fc2_wmma(const _Float16* __restrict__ s3,    // [1280][128]
                         const _Float16* __restrict__ wfc2h, // [16][128]
                         float* __restrict__ z4) {           // [1280][16]
  int wave = threadIdx.x >> 5;
  int lane = threadIdx.x & 31;
  int mt = blockIdx.x * 4 + wave;   // 0..79
  int mlo = lane & 15, half = lane >> 4, ko = half * 16;
  const _Float16* arow = s3 + (size_t)(mt * 16 + mlo) * 128;
  const _Float16* brow = wfc2h + (size_t)mlo * 128;
  v8f acc = zero_v8f();
#pragma unroll
  for (int kc = 0; kc < 4; ++kc) {
    int kb = kc * 32;
    FragH af;
    af.h[0] = *(const v8h*)(arow + kb + ko);
    af.h[1] = *(const v8h*)(arow + kb + 8 + ko);
    FragH bf;
    bf.v = *(const v16h*)(brow + kb + half * 16);
    acc = __builtin_amdgcn_wmma_f32_16x16x32_f16(
        false, af.v, false, bf.v, (short)0, acc, false, false);
  }
#pragma unroll
  for (int r = 0; r < 8; ++r)
    z4[(size_t)(mt * 16 + half * 8 + r) * 16 + mlo] = acc[r];
}

// ---------------------------------------------------------------------------
// Final LIF + spike-rate.  d_out = [rate 64x10][spikes 64x20x10].
// ---------------------------------------------------------------------------
__global__ void lif_out(const float* __restrict__ z4, float* __restrict__ out) {
  int id = blockIdx.x * blockDim.x + threadIdx.x;
  if (id >= 640) return;
  int j = id % 10;
  int b = id / 10;
  float* rate   = out;
  float* spikes = out + 640;
  float v = 0.f, r = 0.f;
  for (int t = 0; t < TSTEP; ++t) {
    float x = z4[(size_t)(b * TSTEP + t) * 16 + j];
    v = v * LBETA + x;
    float s = (v >= 1.f) ? 1.f : 0.f;
    v -= s;
    spikes[b * 200 + t * 10 + j] = s;
    r += s;
  }
  rate[b * 10 + j] = r * (1.f / TSTEP);
}

// ---------------------------------------------------------------------------
extern "C" void kernel_launch(void* const* d_in, const int* in_sizes, int n_in,
                              void* d_out, int out_size, void* d_ws, size_t ws_size,
                              hipStream_t stream) {
  (void)in_sizes; (void)n_in; (void)out_size; (void)ws_size;
  const float* x    = (const float*)d_in[0];
  const float* w1   = (const float*)d_in[1];
  const float* g1   = (const float*)d_in[2];
  const float* b1   = (const float*)d_in[3];
  const float* w2   = (const float*)d_in[4];
  const float* g2   = (const float*)d_in[5];
  const float* b2   = (const float*)d_in[6];
  const float* wfc1 = (const float*)d_in[7];
  const float* wfc2 = (const float*)d_in[8];

  char* ws = (char*)d_ws;
  _Float16* z1     = (_Float16*)(ws + OFF_Z1);
  _Float16* z2     = (_Float16*)(ws + OFF_Z2);
  float*    z3     = (float*)(ws + OFF_Z3);
  float*    z4     = (float*)(ws + OFF_Z4);
  _Float16* a1     = (_Float16*)(ws + OFF_A1);
  _Float16* a2     = (_Float16*)(ws + OFF_A2);
  _Float16* s3     = (_Float16*)(ws + OFF_S3);
  _Float16* w2h    = (_Float16*)(ws + OFF_W2H);
  _Float16* wfc1h  = (_Float16*)(ws + OFF_WFC1H);
  _Float16* wfc2h  = (_Float16*)(ws + OFF_WFC2H);
  float*    stats1 = (float*)(ws + OFF_STATS);        // 64 floats
  float*    stats2 = (float*)(ws + OFF_STATS + 256);  // 128 floats

  hipMemsetAsync(ws + OFF_STATS, 0, 768, stream);
  hipMemsetAsync(ws + OFF_A1, 0, A1_BYTES, stream);   // zero halo for conv2

  repack_weights<<<1776, 256, 0, stream>>>(w2, wfc1, wfc2, w2h, wfc1h, wfc2h);

  conv1_direct<<<125440, 256, 0, stream>>>(x, w1, z1);
  bn_stats<<<512, 256, 0, stream>>>(z1, BTT * 784, 32, stats1);
  lif_pool<<<1568, 256, 0, stream>>>(z1, stats1, g1, b1, a1,
                                     28, 32, 31, 5, 1.f / (BTT * 784.f),
                                     18, 2);

  conv2_wmma<<<4160, 128, 0, stream>>>(a1, w2h, z2);
  bn_stats<<<512, 256, 0, stream>>>(z2, BTT * 196, 64, stats2);
  lif_pool<<<784, 256, 0, stream>>>(z2, stats2, g2, b2, a2,
                                    14, 64, 63, 6, 1.f / (BTT * 196.f),
                                    7, 0);

  fc1_wmma<<<160, 128, 0, stream>>>(a2, wfc1h, z3);
  lif_fc1<<<32, 256, 0, stream>>>(z3, s3);
  fc2_wmma<<<20, 128, 0, stream>>>(s3, wfc2h, z4);
  lif_out<<<3, 256, 0, stream>>>(z4, (float*)d_out);
}
